// SimpleRecurrentPredictor_52450140619194
// MI455X (gfx1250) — compile-verified, matrix operand order checked by reading, not run on hardware
//
#include <hip/hip_runtime.h>
#include <hip/hip_bf16.h>
#include <stdint.h>

// ---------------------------------------------------------------------------
// GRU recurrent predictor for MI455X (gfx1250), persistent-kernel style.
//   B=64, S=256, IN=512, H=1024, O=512, K=4  ->  T = S*K = 1024 serial ticks.
// 64 persistent WGs, hidden dim sliced 16/WG, weights + bf16 h in 300KB LDS,
// v_wmma_f32_16x16x32_bf16 for all GEMMs, bf16 h exchange via
// global_load_async_to_lds_b128 (ASYNCcnt), one atomic grid barrier per tick.
// ---------------------------------------------------------------------------

#define BB    64      // batch
#define SS    256     // seq len
#define INF   512     // input features
#define HH    1024    // hidden
#define OO    512     // output
#define KT    4       // forced ticks per step
#define TT    (SS*KT) // 1024 total ticks
#define NWG   64
#define SLICE (HH/NWG)    // 16 hidden units per WG
#define GC    (3*SLICE)   // 48 gate columns per WG
#define OSL   (OO/NWG)    // 8 output cols per WG
#define THREADS 128       // 4 waves of 32

typedef __attribute__((ext_vector_type(16))) __bf16 v16bf;
typedef __attribute__((ext_vector_type(8)))  float  v8f;

struct alignas(16) U4 { unsigned x, y, z, w; };
union Frag { v16bf v; U4 q[2]; };

// f32 -> bf16 (round-to-nearest-even)
__device__ __forceinline__ unsigned short f2bf(float x) {
    unsigned b = __builtin_bit_cast(unsigned, x);
    b += 0x7FFFu + ((b >> 16) & 1u);
    return (unsigned short)(b >> 16);
}
__device__ __forceinline__ unsigned pack2(float lo, float hi) {
    return (unsigned)f2bf(lo) | ((unsigned)f2bf(hi) << 16);
}
__device__ __forceinline__ float bf2f(unsigned short u) {
    return __builtin_bit_cast(float, (unsigned)u << 16);
}

// 16-bit A/B fragment per ISA 7.12.2: lane holds row (M or N) = lane&15,
// half = lane>>4 -> K in {0..7,16..23} vs {8..15,24..31}. Two aligned 16B loads.
__device__ __forceinline__ v16bf load_bf_frag(const unsigned short* rowbase,
                                              int k0, int half) {
    Frag f;
    f.q[0] = *(const U4*)(rowbase + k0 + half * 8);
    f.q[1] = *(const U4*)(rowbase + k0 + 16 + half * 8);
    return f.v;
}

__device__ __forceinline__ v8f wmma_bf16(v16bf a, v16bf b, v8f c) {
    return __builtin_amdgcn_wmma_f32_16x16x32_bf16(
        false, a, false, b, (short)0, c, false, false);
}

__device__ __forceinline__ float sigmoidf_(float x) {
    return 1.0f / (1.0f + __expf(-x));
}

// CDNA5 async memory->LDS copy (no VGPR round trip, tracked by ASYNCcnt)
__device__ __forceinline__ void async_copy_b128(unsigned lds_off, const void* gaddr) {
    asm volatile("global_load_async_to_lds_b128 %0, %1, off"
                 :: "v"(lds_off), "v"(gaddr) : "memory");
}
__device__ __forceinline__ void wait_async0() {
    asm volatile("s_wait_asynccnt 0x0" ::: "memory");
}

// LDS layout (307200 B, fits 320KB WGP LDS):
//   w_hh : GC  x HH  bf16 (96 KB) | w_ih : GC x INF bf16 (48 KB)
//   w_fc : OSL x HH  bf16 (16 KB) | h    : BB x HH  bf16 (128 KB)
//   gi   : BB  x GC  f32  (12 KB)
#define SMEM_BYTES ((size_t)((GC*HH + GC*INF + OSL*HH + BB*HH) * 2 + BB*GC * 4))

__global__ __launch_bounds__(THREADS)
void gru_persistent(const unsigned short* __restrict__ x_bf,  // [B,S,IN] bf16
                    const float* __restrict__ seq_mask,
                    const float* __restrict__ W_ih,
                    const float* __restrict__ W_hh,
                    const float* __restrict__ b_ih,
                    const float* __restrict__ b_hh,
                    const float* __restrict__ W_fc,
                    const float* __restrict__ b_fc,
                    float* __restrict__ out,              // logits [B,T,O]
                    unsigned short* __restrict__ hbuf0,   // [B,H] bf16 dbl buf
                    unsigned short* __restrict__ hbuf1,
                    int* __restrict__ bar) {
    extern __shared__ unsigned char smem[];
    unsigned short* w_hh_lds = (unsigned short*)smem;              // GC*HH
    unsigned short* w_ih_lds = w_hh_lds + GC * HH;                 // GC*INF
    unsigned short* w_fc_lds = w_ih_lds + GC * INF;                // OSL*HH
    unsigned short* h_lds    = w_fc_lds + OSL * HH;                // BB*HH
    float*          gi_lds   = (float*)(h_lds + BB * HH);          // BB*GC

    const int wg   = blockIdx.x;
    const int tid  = threadIdx.x;
    const int lane = tid & 31;
    const int wid  = tid >> 5;        // wave id == batch M-tile (0..3)
    const int n    = lane & 15;
    const int half = lane >> 4;

    // ---- one-time weight staging into LDS (fp32 -> bf16) ----
    for (int i2 = tid * 2; i2 < GC * HH; i2 += THREADS * 2) {
        int gc = i2 / HH, k = i2 - gc * HH;
        int grow = (gc / SLICE) * HH + wg * SLICE + (gc % SLICE);
        const float* src = W_hh + (size_t)grow * HH + k;
        *(unsigned*)(w_hh_lds + i2) = pack2(src[0], src[1]);
    }
    for (int i2 = tid * 2; i2 < GC * INF; i2 += THREADS * 2) {
        int gc = i2 / INF, k = i2 - gc * INF;
        int grow = (gc / SLICE) * HH + wg * SLICE + (gc % SLICE);
        const float* src = W_ih + (size_t)grow * INF + k;
        *(unsigned*)(w_ih_lds + i2) = pack2(src[0], src[1]);
    }
    for (int i2 = tid * 2; i2 < OSL * HH; i2 += THREADS * 2) {
        int o = i2 / HH, k = i2 - o * HH;
        const float* src = W_fc + (size_t)(wg * OSL + o) * HH + k;
        *(unsigned*)(w_fc_lds + i2) = pack2(src[0], src[1]);
    }
    // initial h = 0 directly in LDS
    for (int i = tid; i < (BB * HH) / 2; i += THREADS)
        ((unsigned*)h_lds)[i] = 0u;
    __syncthreads();

    const int jg = wg * SLICE + n;
    const float bhr = b_hh[jg];
    const float bhz = b_hh[HH + jg];
    const float bhn = b_hh[2 * HH + jg];

    const unsigned h_lds_base = (unsigned)(size_t)(void*)h_lds; // LDS byte addr

    int p = 0;
    unsigned short* hb[2] = { hbuf0, hbuf1 };

    for (int t = 0; t < TT; ++t) {
        unsigned short* hnext = hb[1 - p];
        const int s = t >> 2;

        // ---- gi = x_t @ W_ih^T + b_ih, once per timestep (bf16 x) ----
        if ((t & (KT - 1)) == 0) {
            const unsigned short* arow =
                x_bf + (size_t)(wid * 16 + n) * (SS * INF) + (size_t)s * INF;
            __builtin_prefetch(arow + INF, 0, 1);  // next timestep's row
            const unsigned short* wr0 = w_ih_lds + (0 * SLICE + n) * INF;
            const unsigned short* wr1 = w_ih_lds + (1 * SLICE + n) * INF;
            const unsigned short* wr2 = w_ih_lds + (2 * SLICE + n) * INF;
            v8f a0 = {0.f,0.f,0.f,0.f,0.f,0.f,0.f,0.f}, a1 = a0, a2 = a0;
            for (int k0 = 0; k0 < INF; k0 += 32) {
                v16bf a = load_bf_frag(arow, k0, half);
                a0 = wmma_bf16(a, load_bf_frag(wr0, k0, half), a0);
                a1 = wmma_bf16(a, load_bf_frag(wr1, k0, half), a1);
                a2 = wmma_bf16(a, load_bf_frag(wr2, k0, half), a2);
            }
            float bi0 = b_ih[jg], bi1 = b_ih[HH + jg], bi2 = b_ih[2 * HH + jg];
#pragma unroll
            for (int r = 0; r < 8; ++r) {
                int bb = wid * 16 + r + 8 * half;   // D row per ISA layout
                gi_lds[bb * GC + n]             = a0[r] + bi0;
                gi_lds[bb * GC + SLICE + n]     = a1[r] + bi1;
                gi_lds[bb * GC + 2 * SLICE + n] = a2[r] + bi2;
            }
            __syncthreads();
        }

        // ---- gh = h @ W_hh^T : one A fragment feeds all 3 gate WMMAs ----
        const unsigned short* hrow = h_lds + (wid * 16 + n) * HH;
        {
            const unsigned short* wr0 = w_hh_lds + (0 * SLICE + n) * HH;
            const unsigned short* wr1 = w_hh_lds + (1 * SLICE + n) * HH;
            const unsigned short* wr2 = w_hh_lds + (2 * SLICE + n) * HH;
            v8f g0 = {0.f,0.f,0.f,0.f,0.f,0.f,0.f,0.f}, g1 = g0, g2 = g0;
            for (int k0 = 0; k0 < HH; k0 += 32) {
                v16bf a = load_bf_frag(hrow, k0, half);
                g0 = wmma_bf16(a, load_bf_frag(wr0, k0, half), g0);
                g1 = wmma_bf16(a, load_bf_frag(wr1, k0, half), g1);
                g2 = wmma_bf16(a, load_bf_frag(wr2, k0, half), g2);
            }

            // ---- gate math; h_new stored to global as bf16 ----
#pragma unroll
            for (int r = 0; r < 8; ++r) {
                int bb = wid * 16 + r + 8 * half;
                float ir  = gi_lds[bb * GC + n];
                float iz  = gi_lds[bb * GC + SLICE + n];
                float inn = gi_lds[bb * GC + 2 * SLICE + n];
                float rg = sigmoidf_(ir + g0[r] + bhr);
                float zg = sigmoidf_(iz + g1[r] + bhz);
                float ng = tanhf(inn + rg * (g2[r] + bhn));
                float hp = bf2f(h_lds[bb * HH + jg]);
                hnext[bb * HH + jg] = f2bf((1.0f - zg) * ng + zg * hp);
            }
        }

        // ---- grid-wide barrier: everyone's h_new visible ----
        __threadfence();
        __syncthreads();
        if (tid == 0) {
            atomicAdd(bar, 1);
            const int target = NWG * (t + 1);
            while (*(volatile int*)bar < target) __builtin_amdgcn_s_sleep(2);
        }
        __syncthreads();
        __threadfence();

        // ---- restage full h_new (bf16) into LDS via async b128 copies ----
        for (int i8 = tid * 8; i8 < BB * HH; i8 += THREADS * 8)   // 8 bf16 = 16B
            async_copy_b128(h_lds_base + (unsigned)i8 * 2u, hnext + i8);
        wait_async0();
        __syncthreads();

        // ---- logits = h_new @ W_fc^T (8 real N cols; lanes n>=8 run on
        //      duplicated W rows so EXEC stays all-ones, results dropped) ----
        {
            const unsigned short* wrow = w_fc_lds + (n & (OSL - 1)) * HH;
            v8f acc = {0.f,0.f,0.f,0.f,0.f,0.f,0.f,0.f};
            for (int k0 = 0; k0 < HH; k0 += 32) {
                v16bf a = load_bf_frag(hrow, k0, half);
                acc = wmma_bf16(a, load_bf_frag(wrow, k0, half), acc);
            }
            if (n < OSL) {
                int o = wg * OSL + n;
                float bo = b_fc[o];
#pragma unroll
                for (int r = 0; r < 8; ++r) {
                    int bb = wid * 16 + r + 8 * half;
                    float sm = seq_mask[bb * SS + s];
                    out[((size_t)bb * TT + t) * OO + o] = (acc[r] + bo) * sm;
                }
            }
        }

        p ^= 1;
    }
}

// One-time fp32 -> bf16 conversion of x_seq (removes per-timestep cvt VALU)
__global__ void convert_x(const float* __restrict__ x,
                          unsigned short* __restrict__ xb, int n2) {
    int i2 = 2 * (blockIdx.x * blockDim.x + threadIdx.x);
    if (i2 < n2) {
        float2 v = *(const float2*)(x + i2);
        *(unsigned*)(xb + i2) = pack2(v.x, v.y);
    }
}

// y_res = repeat(y_seq, K) and ok mask, as float after the logits region.
__global__ void aux_outputs(const long long* __restrict__ y_seq,  // int64 in ref
                            const float* __restrict__ seq_mask,
                            float* __restrict__ out) {
    int i = blockIdx.x * blockDim.x + threadIdx.x;
    if (i >= BB * TT) return;
    int b = i / TT, t = i - b * TT, s = t >> 2;
    size_t base = (size_t)BB * TT * OO;
    out[base + i]           = (float)y_seq[b * SS + s];
    out[base + BB * TT + i] = (seq_mask[b * SS + s] != 0.0f) ? 1.0f : 0.0f;
}

extern "C" void kernel_launch(void* const* d_in, const int* in_sizes, int n_in,
                              void* d_out, int out_size, void* d_ws, size_t ws_size,
                              hipStream_t stream) {
    const float*     x_seq    = (const float*)d_in[0];
    const long long* y_seq    = (const long long*)d_in[1];
    const float*     seq_mask = (const float*)d_in[2];
    // d_in[3] = forced_num_ticks (== 4, compile-time KT)
    const float* W_ih = (const float*)d_in[4];
    const float* W_hh = (const float*)d_in[5];
    const float* b_ih = (const float*)d_in[6];
    const float* b_hh = (const float*)d_in[7];
    const float* W_fc = (const float*)d_in[8];
    const float* b_fc = (const float*)d_in[9];
    float* out = (float*)d_out;

    // workspace: [0,256): barrier; 2x bf16 h buffers; bf16 copy of x_seq
    char* ws = (char*)d_ws;
    int*            bar   = (int*)ws;
    unsigned short* hbuf0 = (unsigned short*)(ws + 256);
    unsigned short* hbuf1 = hbuf0 + BB * HH;
    unsigned short* x_bf  = hbuf1 + BB * HH;    // B*S*IN bf16 = 16 MB

    hipMemsetAsync(ws, 0, 256, stream);         // zero barrier counter

    const int n2 = BB * SS * INF;
    convert_x<<<(n2 / 2 + 255) / 256, 256, 0, stream>>>(x_seq, x_bf, n2);

    gru_persistent<<<NWG, THREADS, SMEM_BYTES, stream>>>(
        x_bf, seq_mask, W_ih, W_hh, b_ih, b_hh, W_fc, b_fc,
        out, hbuf0, hbuf1, bar);

    aux_outputs<<<(BB * TT + 255) / 256, 256, 0, stream>>>(y_seq, seq_mask, out);
}
